// MultiHeadLatentAttention_74904229642374
// MI455X (gfx1250) — matrix operand order repaired and checked by reading, not compiled, not run on hardware
//
#include <hip/hip_runtime.h>
#include <stdint.h>

// ---------------- problem constants ----------------
#define B_  2
#define S_  2048
#define D_  2048
#define H_  16
#define DK_ 128
#define DL_ 512
#define DR_ 64
#define EPS_ 1e-6f

typedef __attribute__((ext_vector_type(16))) __bf16 v16bf;
typedef __attribute__((ext_vector_type(8)))  float  v8f;

union Frag { uint32_t u[8]; v16bf v; };

#define WMMA_BF16(a, b, c) \
  __builtin_amdgcn_wmma_f32_16x16x32_bf16(false, (a), false, (b), (short)0, (c), false, false)

__device__ __forceinline__ unsigned short f2bf(float f) {
  uint32_t u = __builtin_bit_cast(uint32_t, f);
  u += 0x7fffu + ((u >> 16) & 1u);   // round to nearest even
  return (unsigned short)(u >> 16);
}

__device__ __forceinline__ v8f zero8() {
  v8f z;
  #pragma unroll
  for (int i = 0; i < 8; ++i) z[i] = 0.0f;
  return z;
}

// A-fragment (16x32 bf16): lane holds row lane%16; reg j holds K pair:
// j<4 : K = half*8 + 2j ; j>=4 : K = 16 + half*8 + 2(j-4)
__device__ __forceinline__ int a_kidx(int j, int half16) {
  return (j < 4) ? (half16 * 8 + 2 * j) : (16 + half16 * 8 + 2 * (j - 4));
}

// CDNA5 async per-lane global->LDS DMA of 16 bytes (ASYNCcnt-tracked).
__device__ __forceinline__ void async_b128(unsigned short* lds, const unsigned short* g) {
  asm volatile("global_load_async_to_lds_b128 %0, %1, off"
               :
               : "v"((uint32_t)(uintptr_t)lds), "v"((uint64_t)(uintptr_t)g)
               : "memory");
}
__device__ __forceinline__ void wait_async0() {
  asm volatile("s_wait_asynccnt 0" ::: "memory");
}
// Wait until only the 4 most recently issued async loads (next tile) are in flight.
__device__ __forceinline__ void wait_async4() {
  asm volatile("s_wait_asynccnt 4" ::: "memory");
}

// ---------------- f32 -> bf16 convert ----------------
__global__ __launch_bounds__(256) void k_cvt(const float* __restrict__ in,
                                             unsigned short* __restrict__ out, long n) {
  long i = (long)blockIdx.x * blockDim.x + threadIdx.x;
  long stride = (long)gridDim.x * blockDim.x;
  for (; i < n; i += stride) out[i] = f2bf(in[i]);
}

// ---------------- f32 [K][N] -> bf16 transposed [N][K] ----------------
__global__ __launch_bounds__(256) void k_cvt_t(const float* __restrict__ in,
                                               unsigned short* __restrict__ out,
                                               int K, int N) {
  __shared__ float tile[32][33];
  const int k0 = blockIdx.x * 32, n0 = blockIdx.y * 32;
  const int tx = threadIdx.x & 31, ty = threadIdx.x >> 5;   // ty: 0..7
  #pragma unroll
  for (int i = ty; i < 32; i += 8) tile[i][tx] = in[(long)(k0 + i) * N + n0 + tx];
  __syncthreads();
  #pragma unroll
  for (int i = ty; i < 32; i += 8) out[(long)(n0 + i) * K + k0 + tx] = f2bf(tile[tx][i]);
}

// ---------------- generic bf16 WMMA GEMM: C[M,N] = A[M,K] @ Bt[N,K]^T ----------------
// A row-major bf16 [M,K]; Bt row-major bf16 [N,K] (i.e. B transposed); C f32 [M,N].
// Double-buffered LDS, async global->LDS staging.
#define BM 128
#define BN 128
#define BK 32
#define LP 8   // LDS row pad (halfwords): stride 40 hw = 80 B -> conflict-free, 16B aligned

__global__ __launch_bounds__(256) void k_gemm_bf16(const unsigned short* __restrict__ A,
                                                   const unsigned short* __restrict__ Bt,
                                                   float* __restrict__ C,
                                                   int M, int N, int K) {
  __shared__ unsigned short As[2][BM][BK + LP];
  __shared__ unsigned short Bs[2][BN][BK + LP];

  const int tid    = threadIdx.x;
  const int lane   = tid & 31;
  const int wave   = tid >> 5;          // 0..7
  const int half16 = lane >> 4;         // 0|1
  const int l16    = lane & 15;

  const int m0 = blockIdx.x * BM;
  const int n0 = blockIdx.y * BN;

  // wave tile: 32(M) x 64(N); waves arranged 4(M) x 2(N)
  const int wm = (wave & 3) * 32;
  const int wn = (wave >> 2) * 64;

  v8f acc[2][4];
  #pragma unroll
  for (int sm = 0; sm < 2; ++sm)
    #pragma unroll
    for (int sn = 0; sn < 4; ++sn) acc[sm][sn] = zero8();

  const int sr = tid >> 1, sseg = (tid & 1) * 16;   // staging coords (both tiles 128x32)
  const unsigned short* arow = A  + (long)(m0 + sr) * K + sseg;
  const unsigned short* brow = Bt + (long)(n0 + sr) * K + sseg;

  auto stage = [&](int buf, int k0) {
    async_b128(&As[buf][sr][sseg],     arow + k0);
    async_b128(&As[buf][sr][sseg + 8], arow + k0 + 8);
    async_b128(&Bs[buf][sr][sseg],     brow + k0);
    async_b128(&Bs[buf][sr][sseg + 8], brow + k0 + 8);
  };

  stage(0, 0);
  int buf = 0;
  for (int k0 = 0; k0 < K; k0 += BK) {
    const bool more = (k0 + BK < K);
    if (more) stage(buf ^ 1, k0 + BK);   // overlap next-tile DMA with this tile's WMMAs
    if (more) wait_async4(); else wait_async0();
    __syncthreads();

    Frag afr[2], bfr[4];
    #pragma unroll
    for (int sm = 0; sm < 2; ++sm) {
      const int row = wm + sm * 16 + l16;
      #pragma unroll
      for (int j = 0; j < 8; ++j)
        afr[sm].u[j] = *(const uint32_t*)&As[buf][row][a_kidx(j, half16)];
    }
    #pragma unroll
    for (int sn = 0; sn < 4; ++sn) {
      const int col = wn + sn * 16 + l16;
      #pragma unroll
      for (int j = 0; j < 8; ++j)        // same K pairs as A fragment (a_kidx)
        bfr[sn].u[j] = *(const uint32_t*)&Bs[buf][col][a_kidx(j, half16)];
    }
    #pragma unroll
    for (int sm = 0; sm < 2; ++sm)
      #pragma unroll
      for (int sn = 0; sn < 4; ++sn)
        acc[sm][sn] = WMMA_BF16(afr[sm].v, bfr[sn].v, acc[sm][sn]);

    __syncthreads();   // protect buf^1 (restaged next iteration) until all reads done
    buf ^= 1;
  }

  // C layout: reg r -> row r + 8*half16, col l16
  #pragma unroll
  for (int sm = 0; sm < 2; ++sm)
    #pragma unroll
    for (int sn = 0; sn < 4; ++sn)
      #pragma unroll
      for (int r = 0; r < 8; ++r) {
        const int m = wm + sm * 16 + r + half16 * 8;
        const int n = wn + sn * 16 + l16;
        C[(long)(m0 + m) * N + (n0 + n)] = acc[sm][sn][r];
      }
}

// ---------------- q epilogue: rmsnorm(DK) then RoPE on [0,DR) ----------------
// in : qf [B*S, H*DK] f32 ; out: qb [B,H,S,DK] bf16
__global__ __launch_bounds__(128) void k_q_post(const float* __restrict__ qf,
                                                const float* __restrict__ qw,
                                                unsigned short* __restrict__ qb) {
  const int idx = blockIdx.x;            // (b*S + s)*H + h
  const int h = idx % H_;
  const int bs = idx / H_;
  const int s = bs % S_;
  const int b = bs / S_;
  const int t = threadIdx.x;             // dk

  __shared__ float red[4];
  __shared__ float vals[DK_];

  float x = qf[(long)bs * (H_ * DK_) + h * DK_ + t];
  float ss = x * x;
  #pragma unroll
  for (int off = 16; off; off >>= 1) ss += __shfl_xor(ss, off);
  if ((t & 31) == 0) red[t >> 5] = ss;
  __syncthreads();
  const float inv = rsqrtf((red[0] + red[1] + red[2] + red[3]) * (1.0f / DK_) + EPS_);
  const float xn = x * inv * qw[t];
  vals[t] = xn;
  __syncthreads();

  float out = xn;
  if (t < DR_) {
    const int j = t & 31;                                    // half of DR
    const float fr = (j < 16) ? __powf(1.0e-4f, (float)j * (1.0f / 15.0f)) : 0.0f;
    const float th = fr * (float)s;
    const float cs = __cosf(th), sn = __sinf(th);
    const float x1 = vals[j], x2 = vals[j + 32];
    out = (t < 32) ? (x1 * cs + x2 * sn) : (-x1 * sn + x2 * cs);
  }
  qb[(((long)b * H_ + h) * S_ + s) * DK_ + t] = f2bf(out);
}

// ---------------- k epilogue: RoPE(k_rope) ++ k_nope, then rmsnorm(DK) ----------------
__global__ __launch_bounds__(128) void k_k_post(const float* __restrict__ krf,  // [B*S, H*DR]
                                                const float* __restrict__ knf,  // [B*S, H*(DK-DR)]
                                                const float* __restrict__ kw,
                                                unsigned short* __restrict__ kb) {
  const int idx = blockIdx.x;
  const int h = idx % H_;
  const int bs = idx / H_;
  const int s = bs % S_;
  const int b = bs / S_;
  const int t = threadIdx.x;

  __shared__ float red[4];
  __shared__ float vals[DK_];

  float raw;
  if (t < DR_) raw = krf[(long)bs * (H_ * DR_) + h * DR_ + t];
  else         raw = knf[(long)bs * (H_ * (DK_ - DR_)) + h * (DK_ - DR_) + (t - DR_)];
  vals[t] = raw;
  __syncthreads();

  float kv = raw;
  if (t < DR_) {
    const int j = t & 31;
    const float fr = (j < 16) ? __powf(1.0e-4f, (float)j * (1.0f / 15.0f)) : 0.0f;
    const float th = fr * (float)s;
    const float cs = __cosf(th), sn = __sinf(th);
    const float x1 = vals[j], x2 = vals[j + 32];
    kv = (t < 32) ? (x1 * cs + x2 * sn) : (-x1 * sn + x2 * cs);
  }
  float ss = kv * kv;
  #pragma unroll
  for (int off = 16; off; off >>= 1) ss += __shfl_xor(ss, off);
  __syncthreads();
  if ((t & 31) == 0) red[t >> 5] = ss;
  __syncthreads();
  const float inv = rsqrtf((red[0] + red[1] + red[2] + red[3]) * (1.0f / DK_) + EPS_);
  kb[(((long)b * H_ + h) * S_ + s) * DK_ + t] = f2bf(kv * inv * kw[t]);
}

// ---------------- v: [B,S,H,DK] f32 -> [B,H,DK,S] bf16 (transposed for PV WMMA) ------
__global__ __launch_bounds__(256) void k_v_post(const float* __restrict__ vf,
                                                unsigned short* __restrict__ vb) {
  long i = (long)blockIdx.x * blockDim.x + threadIdx.x;     // index in OUTPUT order
  const long n = (long)B_ * H_ * DK_ * S_;
  const long stride = (long)gridDim.x * blockDim.x;
  for (; i < n; i += stride) {
    const int s = (int)(i % S_);
    long r = i / S_;
    const int dk = (int)(r % DK_);
    r /= DK_;
    const int h = (int)(r % H_);
    const int b = (int)(r / H_);
    vb[i] = f2bf(vf[(((long)b * S_ + s) * H_ + h) * DK_ + dk]);
  }
}

// ---------------- flash attention (causal, online softmax, WMMA) ----------------
// q,k : [B*H, S, DK] bf16 ; v : [B*H, DK, S] bf16 ; o : [B, S, H*DK] bf16
// Double-buffered K/V tiles, async staging.
#define QB 128
#define KT 32

__global__ __launch_bounds__(256) void k_flash(const unsigned short* __restrict__ q,
                                               const unsigned short* __restrict__ k,
                                               const unsigned short* __restrict__ v,
                                               unsigned short* __restrict__ o) {
  __shared__ unsigned short Ks[2][KT][DK_ + LP];   // [key][dk]  row 272B
  __shared__ unsigned short Vs[2][DK_][KT + LP];   // [dk][key]  row 80B
  __shared__ unsigned short Ps[8][16][KT + LP];    // per-wave P scratch, row 80B

  const int bh = blockIdx.x;
  const int b = bh / H_, h = bh % H_;
  const int q0 = blockIdx.y * QB;
  const int tid = threadIdx.x, lane = tid & 31, wave = tid >> 5;
  const int half16 = lane >> 4, l16 = lane & 15;
  const int qt0 = q0 + wave * 16;

  // Q fragments: 4 chunks of K=32 over DK=128; held for whole key loop
  Frag qf[4];
  {
    const unsigned short* qrow = q + ((long)bh * S_ + qt0 + l16) * DK_;
    #pragma unroll
    for (int ch = 0; ch < 4; ++ch)
      #pragma unroll
      for (int j = 0; j < 8; ++j)
        qf[ch].u[j] = *(const uint32_t*)(qrow + ch * 32 + a_kidx(j, half16));
  }

  float mrow[8], lrow[8];
  v8f oacc[8];
  #pragma unroll
  for (int r = 0; r < 8; ++r) { mrow[r] = -1e30f; lrow[r] = 0.0f; }
  #pragma unroll
  for (int d = 0; d < 8; ++d) oacc[d] = zero8();

  const float scale = 0.08838834764831845f;   // 1/sqrt(DK)
  const int nkt = (q0 + QB) / KT;
  const int qmaxw = qt0 + 15;

  const int kr = tid >> 3, kseg = (tid & 7) * 16;   // K tile staging (32 x 128)
  const int vr = tid >> 1, vseg = (tid & 1) * 16;   // V tile staging (128 x 32)
  const unsigned short* krow = k + ((long)bh * S_ + kr) * DK_ + kseg;
  const unsigned short* vrow = v + ((long)bh * DK_ + vr) * S_ + vseg;

  auto stage = [&](int buf, int kb) {
    async_b128(&Ks[buf][kr][kseg],     krow + (long)kb * DK_);
    async_b128(&Ks[buf][kr][kseg + 8], krow + (long)kb * DK_ + 8);
    async_b128(&Vs[buf][vr][vseg],     vrow + kb);
    async_b128(&Vs[buf][vr][vseg + 8], vrow + kb + 8);
  };

  stage(0, 0);
  int buf = 0;
  for (int kt = 0; kt < nkt; ++kt) {
    const int kb = kt * KT;
    const bool more = (kt + 1 < nkt);
    if (more) stage(buf ^ 1, kb + KT);   // overlap next K/V tile DMA with compute
    if (more) wait_async4(); else wait_async0();
    __syncthreads();

    if (kb <= qmaxw) {  // wave-uniform causal skip
      // ---- scores: S = Q(16xDK) . K^T(DKx32), two 16-key column tiles ----
      v8f sc[2];
      #pragma unroll
      for (int n16 = 0; n16 < 2; ++n16) {
        v8f c = zero8();
        const int col = n16 * 16 + l16;   // key within tile
        #pragma unroll
        for (int ch = 0; ch < 4; ++ch) {
          Frag bf;
          #pragma unroll
          for (int j = 0; j < 8; ++j)     // dk pairs in A-fragment order (a_kidx)
            bf.u[j] = *(const uint32_t*)&Ks[buf][col][ch * 32 + a_kidx(j, half16)];
          c = WMMA_BF16(qf[ch].v, bf.v, c);
        }
        sc[n16] = c;
      }

      // ---- online softmax (row = r + 8*half16 across the 16 lanes of a half) ----
      float pr0[8], pr1[8], scl[8];
      #pragma unroll
      for (int r = 0; r < 8; ++r) {
        const int qrow = qt0 + r + half16 * 8;
        float x = sc[0][r] * scale; if (kb + l16 > qrow)      x = -1e30f;
        float y = sc[1][r] * scale; if (kb + 16 + l16 > qrow) y = -1e30f;
        float mx = fmaxf(x, y);
        #pragma unroll
        for (int off = 1; off < 16; off <<= 1) mx = fmaxf(mx, __shfl_xor(mx, off));
        const float mo = mrow[r];
        const float mn = fmaxf(mo, mx);
        const float sr2 = __expf(mo - mn);
        const float p0 = __expf(x - mn);
        const float p1 = __expf(y - mn);
        float rs = p0 + p1;
        #pragma unroll
        for (int off = 1; off < 16; off <<= 1) rs += __shfl_xor(rs, off);
        lrow[r] = lrow[r] * sr2 + rs;
        mrow[r] = mn;
        scl[r] = sr2; pr0[r] = p0; pr1[r] = p1;
      }
      #pragma unroll
      for (int d = 0; d < 8; ++d)
        #pragma unroll
        for (int r = 0; r < 8; ++r) oacc[d][r] *= scl[r];

      // ---- rebuild P(16x32) into A-fragment layout via per-wave LDS ----
      #pragma unroll
      for (int r = 0; r < 8; ++r) {
        Ps[wave][r + half16 * 8][l16]      = f2bf(pr0[r]);
        Ps[wave][r + half16 * 8][16 + l16] = f2bf(pr1[r]);
      }
      asm volatile("s_wait_dscnt 0" ::: "memory");   // in-wave LDS RAW ordering
      Frag pf;
      #pragma unroll
      for (int j = 0; j < 8; ++j)
        pf.u[j] = *(const uint32_t*)&Ps[wave][l16][a_kidx(j, half16)];

      // ---- O += P(16x32) . V(32xDK) ----
      #pragma unroll
      for (int d = 0; d < 8; ++d) {
        Frag vfr;
        const int col = d * 16 + l16;     // dk column -> Vs row
        #pragma unroll
        for (int j = 0; j < 8; ++j)       // key pairs in A-fragment order (a_kidx)
          vfr.u[j] = *(const uint32_t*)&Vs[buf][col][a_kidx(j, half16)];
        oacc[d] = WMMA_BF16(pf.v, vfr.v, oacc[d]);
      }
    }
    __syncthreads();
    buf ^= 1;
  }

  // ---- epilogue: O / l -> [B, S, H*DK] bf16 ----
  #pragma unroll
  for (int r = 0; r < 8; ++r) {
    const int qrow = qt0 + r + half16 * 8;
    const float invl = 1.0f / lrow[r];
    unsigned short* dst = o + ((long)b * S_ + qrow) * (H_ * DK_) + h * DK_;
    #pragma unroll
    for (int d = 0; d < 8; ++d) dst[d * 16 + l16] = f2bf(oacc[d][r] * invl);
  }
}

// ---------------- host launcher ----------------
extern "C" void kernel_launch(void* const* d_in, const int* in_sizes, int n_in,
                              void* d_out, int out_size, void* d_ws, size_t ws_size,
                              hipStream_t stream) {
  const float* x   = (const float*)d_in[0];
  const float* wq  = (const float*)d_in[1];
  const float* wkv = (const float*)d_in[2];
  const float* wkr = (const float*)d_in[3];
  const float* wkn = (const float*)d_in[4];
  const float* wv  = (const float*)d_in[5];
  const float* wo  = (const float*)d_in[6];
  const float* qnw = (const float*)d_in[7];
  const float* knw = (const float*)d_in[8];

  const long XB   = (long)B_ * S_ * D_;          // 8M
  const long WQB  = (long)D_ * H_ * DK_;         // 4M
  const long WKVD = (long)D_ * DL_;              // 1M
  const long WKR  = (long)DL_ * H_ * DR_;        // 512K
  const long WKN  = (long)DL_ * H_ * (DK_ - DR_);// 512K
  const long WVB  = (long)DL_ * H_ * DK_;        // 1M
  const long WOB  = (long)H_ * DK_ * D_;         // 4M
  const long QF   = (long)B_ * S_ * H_ * DK_;    // 8M
  const long LATF = (long)B_ * S_ * DL_;         // 2M
  const long KRF  = (long)B_ * S_ * H_ * DR_;    // 4M
  const long KNF  = (long)B_ * S_ * H_ * (DK_ - DR_);

  char* p = (char*)d_ws;
  auto alloc = [&](size_t bytes) -> void* {
    void* r = (void*)p;
    p += (bytes + 255) & ~(size_t)255;
    return r;
  };
  unsigned short* xb   = (unsigned short*)alloc(XB * 2);
  unsigned short* wqb  = (unsigned short*)alloc(WQB * 2);   // transposed [N][K]
  unsigned short* wkvb = (unsigned short*)alloc(WKVD * 2);  // transposed
  unsigned short* wkrb = (unsigned short*)alloc(WKR * 2);   // transposed
  unsigned short* wknb = (unsigned short*)alloc(WKN * 2);   // transposed
  unsigned short* wvb  = (unsigned short*)alloc(WVB * 2);   // transposed
  unsigned short* wob  = (unsigned short*)alloc(WOB * 2);   // transposed
  unsigned short* latb = (unsigned short*)alloc(LATF * 2);
  unsigned short* qb2  = (unsigned short*)alloc(QF * 2);
  unsigned short* kb2  = (unsigned short*)alloc(QF * 2);
  unsigned short* vb2  = (unsigned short*)alloc(QF * 2);    // [B,H,DK,S]
  unsigned short* aob  = (unsigned short*)alloc(QF * 2);
  float* qf   = (float*)alloc(QF * 4);
  float* latf = (float*)alloc(LATF * 4);
  float* vfw  = (float*)alloc(QF * 4);
  float* krf  = (float*)alloc(KRF * 4);
  float* knf  = (float*)alloc(KNF * 4);

  auto cvt = [&](const float* in, unsigned short* out, long n) {
    k_cvt<<<4096, 256, 0, stream>>>(in, out, n);
  };
  auto cvtT = [&](const float* in, unsigned short* out, int K, int N) {
    k_cvt_t<<<dim3(K / 32, N / 32), 256, 0, stream>>>(in, out, K, N);
  };
  cvt(x, xb, XB);
  cvtT(wq,  wqb,  D_,  H_ * DK_);
  cvtT(wkv, wkvb, D_,  DL_);
  cvtT(wkr, wkrb, DL_, H_ * DR_);
  cvtT(wkn, wknb, DL_, H_ * (DK_ - DR_));
  cvtT(wv,  wvb,  DL_, H_ * DK_);
  cvtT(wo,  wob,  H_ * DK_, D_);

  const int M = B_ * S_;   // 4096
  // q = x @ wq  : [4096,2048] x [2048,2048]
  k_gemm_bf16<<<dim3(M / BM, (H_ * DK_) / BN), 256, 0, stream>>>(xb, wqb, qf, M, H_ * DK_, D_);
  // latent = x @ w_kv_down : [4096,2048] x [2048,512]
  k_gemm_bf16<<<dim3(M / BM, DL_ / BN), 256, 0, stream>>>(xb, wkvb, latf, M, DL_, D_);
  cvt(latf, latb, LATF);
  // v = latent @ wv : [4096,512] x [512,2048]
  k_gemm_bf16<<<dim3(M / BM, (H_ * DK_) / BN), 256, 0, stream>>>(latb, wvb, vfw, M, H_ * DK_, DL_);
  // k_rope = latent @ w_k_rope : [4096,512] x [512,1024]
  k_gemm_bf16<<<dim3(M / BM, (H_ * DR_) / BN), 256, 0, stream>>>(latb, wkrb, krf, M, H_ * DR_, DL_);
  // k_nope = latent @ w_k_nope : [4096,512] x [512,1024]
  k_gemm_bf16<<<dim3(M / BM, (H_ * (DK_ - DR_)) / BN), 256, 0, stream>>>(latb, wknb, knf, M, H_ * (DK_ - DR_), DL_);

  k_q_post<<<M * H_, 128, 0, stream>>>(qf, qnw, qb2);
  k_k_post<<<M * H_, 128, 0, stream>>>(krf, knf, knw, kb2);
  k_v_post<<<8192, 256, 0, stream>>>(vfw, vb2);

  k_flash<<<dim3(B_ * H_, S_ / QB), 256, 0, stream>>>(qb2, kb2, vb2, aob);

  // out = attn_out @ wo : [4096,2048] x [2048,2048] -> f32 d_out
  k_gemm_bf16<<<dim3(M / BM, D_ / BN), 256, 0, stream>>>(aob, wob, (float*)d_out, M, D_, H_ * DK_);
}
